// KappaGCN_69758858822203
// MI455X (gfx1250) — matrix operand order repaired and checked by reading, not compiled, not run on hardware
//
#include <hip/hip_runtime.h>
#include <hip/hip_bf16.h>

// ---------------------------------------------------------------------------
// KappaGCN layer for MI455X (gfx1250, wave32, WMMA).
// Pipeline: k1w -> k1 (X@W + mobius scale, bf16 WMMA)
//           -> k2 (A @ [lam*XW | lam-1 | 1], bf16 WMMA, A cvt on the fly,
//                  NT loads for streamed A, deep global_prefetch)
//           -> k3 (per-row hyperbolic ops + per-class scalars, shfl reduce,
//                  deterministic per-block z^2 partials)
//           -> k4a/k4b (class constants, asinh logits)
//           -> k5 (A @ logits, fp32 WMMA 16x16x4, NT A loads)
// ---------------------------------------------------------------------------

typedef __attribute__((ext_vector_type(16))) __bf16       v16bf;
typedef __attribute__((ext_vector_type(8)))  float        v8f;
typedef __attribute__((ext_vector_type(8)))  unsigned int v8u;
typedef __attribute__((ext_vector_type(4)))  float        f4v;
typedef __attribute__((ext_vector_type(2)))  float        v2f;

#define MIN_NORM 1e-15f
#define N_ROWS   8192
#define DIM      256
#define NCLS     16
#define NCOL     272   /* 256 data cols + den + rsum + 14 pad -> 17 N-tiles */
#define K3_BLKS  128

// ---- helpers --------------------------------------------------------------

__device__ __forceinline__ unsigned int packbf(float lo, float hi) {
  // packed RNE fp32->bf16x2; lowers to native packed cvt when available
  __hip_bfloat162 h = __float22bfloat162_rn(make_float2(lo, hi));
  unsigned int u;
  __builtin_memcpy(&u, &h, 4);
  return u;
}
__device__ __forceinline__ unsigned short f2bfs(float f) {
  __hip_bfloat16 h = __float2bfloat16(f);
  unsigned short s;
  __builtin_memcpy(&s, &h, 2);
  return s;
}

__device__ __forceinline__ float artanh_c(float x) {  // clip like reference
  x = fminf(fmaxf(x, -1.0f + 1e-6f), 1.0f - 1e-6f);
  return atanhf(x);
}

__device__ __forceinline__ float wave_sum32(float v) {
  #pragma unroll
  for (int m = 16; m >= 1; m >>= 1) v += __shfl_xor(v, m);
  return v;
}

// ---- k1w: WbT[c][k] = bf16(W[k][c]) --------------------------------------

__global__ void __launch_bounds__(256) k1w_transpose(
    const float* __restrict__ W, unsigned short* __restrict__ WbT) {
  int idx = blockIdx.x * 256 + threadIdx.x;   // idx = c*256 + k
  int c = idx >> 8, k = idx & 255;
  WbT[idx] = f2bfs(W[k * DIM + c]);
}

// ---- k1: MX = X @ W (bf16 WMMA), then mobius_matvec scale + lambda,
//          write Yt (K-major, 272 x 8192 bf16) -------------------------------

__global__ void __launch_bounds__(256) k1_mobius_matvec(
    const float* __restrict__ X, const unsigned short* __restrict__ WbT,
    unsigned short* __restrict__ Yt) {
  const int wave = threadIdx.x >> 5;
  const int lane = threadIdx.x & 31;
  const int hi   = lane >> 4;
  const int ln   = lane & 15;
  const int m0   = blockIdx.x * 128 + wave * 16;

  const float* xrow = X + (long)(m0 + ln) * DIM;

  v8f acc[16];
  #pragma unroll
  for (int t = 0; t < 16; ++t) acc[t] = 0.0f;

  float xsq = 0.0f;

  for (int k0 = 0; k0 < DIM; k0 += 32) {
    // A fragment: 16-bit A 16x32 layout (lane<16: K 0..7,16..23; lane>=16: 8..15,24..31)
    const int o1 = k0 + hi * 8;
    const int o2 = k0 + 16 + hi * 8;
    f4v a0 = *(const f4v*)(xrow + o1);
    f4v a1 = *(const f4v*)(xrow + o1 + 4);
    f4v a2 = *(const f4v*)(xrow + o2);
    f4v a3 = *(const f4v*)(xrow + o2 + 4);
    #pragma unroll
    for (int e = 0; e < 4; ++e)
      xsq += a0[e]*a0[e] + a1[e]*a1[e] + a2[e]*a2[e] + a3[e]*a3[e];
    v8u ua;
    ua[0] = packbf(a0[0], a0[1]); ua[1] = packbf(a0[2], a0[3]);
    ua[2] = packbf(a1[0], a1[1]); ua[3] = packbf(a1[2], a1[3]);
    ua[4] = packbf(a2[0], a2[1]); ua[5] = packbf(a2[2], a2[3]);
    ua[6] = packbf(a3[0], a3[1]); ua[7] = packbf(a3[2], a3[3]);
    v16bf fa = __builtin_bit_cast(v16bf, ua);

    #pragma unroll
    for (int t = 0; t < 16; ++t) {
      // B fragment: lane = column, K 8h..8h+7 and 16+8h..16+8h+7 from WbT row
      const uint4* bp = (const uint4*)(WbT + (t * 16 + ln) * DIM + k0 + hi * 8);
      uint4 b0 = bp[0], b1 = bp[2];
      v8u ub;
      ub[0] = b0.x; ub[1] = b0.y; ub[2] = b0.z; ub[3] = b0.w;
      ub[4] = b1.x; ub[5] = b1.y; ub[6] = b1.z; ub[7] = b1.w;
      v16bf fb = __builtin_bit_cast(v16bf, ub);
      acc[t] = __builtin_amdgcn_wmma_f32_16x16x32_bf16(
          false, fa, false, fb, (short)0, acc[t], false, false);
    }
  }

  // ||X_row||^2: lanes l and l+16 hold disjoint K halves of row m0+(l&15)
  float xn2 = xsq + __shfl_xor(xsq, 16);

  // Epilogue: per output row (C layout: VGPR j = row j (lanes<16) / row 8+j)
  #pragma unroll
  for (int j = 0; j < 8; ++j) {
    float s = 0.0f;
    #pragma unroll
    for (int t = 0; t < 16; ++t) s += acc[t][j] * acc[t][j];
    #pragma unroll
    for (int m = 8; m >= 1; m >>= 1) s += __shfl_xor(s, m);  // within 16-lane half

    const int r = m0 + hi * 8 + j;
    float xn2j = __shfl(xn2, hi * 8 + j);       // row r's ||X||^2
    float Xn  = fmaxf(sqrtf(xn2j), MIN_NORM);
    float MXn = fmaxf(sqrtf(s), MIN_NORM);
    float th  = tanhf(MXn / Xn * artanh_c(Xn)); // ||XW|| = th
    float scale = th / MXn;
    float lam = 2.0f / fmaxf(1.0f - th * th, MIN_NORM);
    float g   = lam * scale;

    #pragma unroll
    for (int t = 0; t < 16; ++t) {
      int c = t * 16 + ln;
      Yt[(long)c * N_ROWS + r] = f2bfs(g * acc[t][j]);
    }
    if (ln == 0) {                               // den / rsum / pad columns
      Yt[(long)256 * N_ROWS + r] = f2bfs(lam - 1.0f);
      Yt[(long)257 * N_ROWS + r] = f2bfs(1.0f);
      #pragma unroll
      for (int c = 258; c < NCOL; ++c) Yt[(long)c * N_ROWS + r] = 0;
    }
  }
}

// ---- k2: OUT272 = A(8192x8192, fp32->bf16 on the fly) @ Yt^T --------------

__global__ void __launch_bounds__(256) k2_aggregate(
    const float* __restrict__ A, const unsigned short* __restrict__ Yt,
    float* __restrict__ OUT) {
  const int wave = threadIdx.x >> 5;
  const int lane = threadIdx.x & 31;
  const int hi   = lane >> 4;
  const int ln   = lane & 15;
  const int m0   = blockIdx.x * 128 + wave * 16;

  const float* arow = A + (long)(m0 + ln) * N_ROWS;

  v8f acc[17];
  #pragma unroll
  for (int t = 0; t < 17; ++t) acc[t] = 0.0f;

  for (int k0 = 0; k0 < N_ROWS; k0 += 32) {
    const int o1 = k0 + hi * 8;
    const int o2 = k0 + 16 + hi * 8;
    // A is streamed once: non-temporal so it never evicts L2-resident Yt/OUT
    f4v a0 = __builtin_nontemporal_load((const f4v*)(arow + o1));
    f4v a1 = __builtin_nontemporal_load((const f4v*)(arow + o1 + 4));
    f4v a2 = __builtin_nontemporal_load((const f4v*)(arow + o2));
    f4v a3 = __builtin_nontemporal_load((const f4v*)(arow + o2 + 4));
    v8u ua;
    ua[0] = packbf(a0[0], a0[1]); ua[1] = packbf(a0[2], a0[3]);
    ua[2] = packbf(a1[0], a1[1]); ua[3] = packbf(a1[2], a1[3]);
    ua[4] = packbf(a2[0], a2[1]); ua[5] = packbf(a2[2], a2[3]);
    ua[6] = packbf(a3[0], a3[1]); ua[7] = packbf(a3[2], a3[3]);
    v16bf fa = __builtin_bit_cast(v16bf, ua);

    // deep prefetch: 4KB ahead on this lane's A stream
    if (k0 + 1024 < N_ROWS) __builtin_prefetch(arow + k0 + 1024 + hi * 8, 0, 0);

    #pragma unroll
    for (int t = 0; t < 17; ++t) {
      const uint4* bp = (const uint4*)(Yt + (t * 16 + ln) * N_ROWS + k0 + hi * 8);
      uint4 b0 = bp[0], b1 = bp[2];
      v8u ub;
      ub[0] = b0.x; ub[1] = b0.y; ub[2] = b0.z; ub[3] = b0.w;
      ub[4] = b1.x; ub[5] = b1.y; ub[6] = b1.z; ub[7] = b1.w;
      v16bf fb = __builtin_bit_cast(v16bf, ub);
      acc[t] = __builtin_amdgcn_wmma_f32_16x16x32_bf16(
          false, fa, false, fb, (short)0, acc[t], false, false);
    }
  }

  #pragma unroll
  for (int t = 0; t < 17; ++t) {
    int n = t * 16 + ln;
    #pragma unroll
    for (int j = 0; j < 8; ++j) {
      int r = m0 + hi * 8 + j;
      OUT[(long)r * NCOL + n] = acc[t][j];
    }
  }
}

// ---- k3: per-row mobius midpoint/scalar-mul/log-relu-exp + class scalars --

__global__ void __launch_bounds__(256) k3_rowops(
    const float* __restrict__ OUT272, const float* __restrict__ p_ks,
    const float* __restrict__ Wl, float* __restrict__ innerW,
    float* __restrict__ zpart) {
  __shared__ float p_lds[NCLS * DIM];
  __shared__ float wl_lds[NCLS * DIM];
  __shared__ float p2s[NCLS];
  __shared__ float pwl[NCLS];
  __shared__ float z2sh[8][NCLS];

  for (int i = threadIdx.x; i < NCLS * DIM; i += 256) {
    p_lds[i] = p_ks[i];                       // (k, c) row-major already
    int k = i & 15, c = i >> 4;               // i = c*16 + k
    wl_lds[k * DIM + c] = Wl[i];              // transpose W_logits
  }
  __syncthreads();
  if (threadIdx.x < NCLS) {
    int k = threadIdx.x;
    float a = 0.0f, b = 0.0f;
    for (int c = 0; c < DIM; ++c) {
      float pv = p_lds[k * DIM + c];
      a += pv * pv;
      b += pv * wl_lds[k * DIM + c];
    }
    p2s[k] = a; pwl[k] = b;
  }
  __syncthreads();

  const int wave = threadIdx.x >> 5;
  const int lane = threadIdx.x & 31;
  const int c0 = lane * 8;

  float z2acc[NCLS];
  #pragma unroll
  for (int k = 0; k < NCLS; ++k) z2acc[k] = 0.0f;

  for (int rr = wave; rr < 64; rr += 8) {
    const int r = blockIdx.x * 64 + rr;
    const float* row = OUT272 + (long)r * NCOL;

    float den = row[256];
    den = (fabsf(den) < MIN_NORM) ? MIN_NORM : den;
    float rsum = row[257];

    float tm[8];
    float tn2 = 0.0f;
    #pragma unroll
    for (int j = 0; j < 8; ++j) {
      float v = row[c0 + j] / den;            // two_mean
      tm[j] = v; tn2 += v * v;
    }
    tn2 = wave_sum32(tn2);
    float tn = fmaxf(sqrtf(tn2), MIN_NORM);
    float s1 = tanhf(0.5f * artanh_c(tn)) / tn;          // mid = s1*tm
    float midn = fmaxf(fabsf(s1) * tn, MIN_NORM);
    float s2 = tanhf(rsum * artanh_c(midn)) / midn;      // AXW = s2*mid
    float axn = fmaxf(fabsf(s2) * midn, MIN_NORM);       // ||AXW||
    float s3 = artanh_c(axn) / axn;                      // logmap0 scale
    float g = s3 * s2 * s1;

    float ru[8];
    float en2 = 0.0f;
    #pragma unroll
    for (int j = 0; j < 8; ++j) {
      float u = fmaxf(g * tm[j], 0.0f);                  // relu(logmap0)
      ru[j] = u; en2 += u * u;
    }
    en2 = wave_sum32(en2);
    float en = fmaxf(sqrtf(en2), MIN_NORM);
    float th = tanhf(en);
    float s4 = th / en;                                  // expmap0 scale
    float H[8];
    #pragma unroll
    for (int j = 0; j < 8; ++j) H[j] = s4 * ru[j];
    float h2 = th * th;                                  // ||H1||^2

    #pragma unroll
    for (int k = 0; k < NCLS; ++k) {
      float ph = 0.0f, hw = 0.0f;
      #pragma unroll
      for (int j = 0; j < 8; ++j) {
        ph += p_lds[k * DIM + c0 + j] * H[j];
        hw += wl_lds[k * DIM + c0 + j] * H[j];
      }
      ph = wave_sum32(ph);
      hw = wave_sum32(hw);
      // Z = mobius_add(-p_k, H1): closed forms over scalars only
      float alpha = 1.0f - 2.0f * ph + h2;               // 1 + 2<x,y> + |y|^2
      float beta  = 1.0f - p2s[k];                       // 1 - |x|^2
      float ds = 1.0f - 2.0f * ph + p2s[k] * h2;
      ds = (fabsf(ds) < MIN_NORM) ? MIN_NORM : ds;
      float inner = (-alpha * pwl[k] + beta * hw) / ds;            // Z . Wl_k
      float z2 = (alpha * alpha * p2s[k] + beta * beta * h2
                  - 2.0f * alpha * beta * ph) / (ds * ds);         // ||Z_row||^2
      z2acc[k] += z2;                                    // deterministic order
      if (lane == 0) innerW[(long)r * NCLS + k] = inner;
    }
  }

  // deterministic block-level reduction of z^2 partials
  if (lane == 0) {
    #pragma unroll
    for (int k = 0; k < NCLS; ++k) z2sh[wave][k] = z2acc[k];
  }
  __syncthreads();
  if (threadIdx.x < NCLS) {
    int k = threadIdx.x;
    float z = 0.0f;
    #pragma unroll
    for (int w = 0; w < 8; ++w) z += z2sh[w][k];         // fixed order
    zpart[blockIdx.x * NCLS + k] = z;
  }
}

// ---- k4a: per-class constants (deterministic z^2 reduce) -------------------

__global__ void k4a_consts(const float* __restrict__ Wl,
                           const float* __restrict__ p_ks,
                           const float* __restrict__ zpart,
                           float* __restrict__ cl, float* __restrict__ cd) {
  int k = threadIdx.x;
  if (k >= NCLS) return;
  float z = 0.0f;
  for (int b = 0; b < K3_BLKS; ++b) z += zpart[b * NCLS + k];  // fixed order
  float a2 = 0.0f, p2 = 0.0f;
  for (int c = 0; c < DIM; ++c) {
    float w = Wl[c * NCLS + k];  a2 += w * w;
    float p = p_ks[k * DIM + c]; p2 += p * p;
  }
  float an  = fmaxf(sqrtf(a2), MIN_NORM);
  float lam = 2.0f / fmaxf(1.0f - p2, MIN_NORM);
  cl[k] = lam * an;                         // lhs
  cd[k] = (1.0f - z) * an;                  // (1 + kappa*z_norm^2)*a_norm
}

// ---- k4b: logits = lhs * asinh(2*inner/denom) -----------------------------

__global__ void __launch_bounds__(256) k4b_logits(
    const float* __restrict__ innerW, const float* __restrict__ cl,
    const float* __restrict__ cd, float* __restrict__ L) {
  int i = blockIdx.x * 256 + threadIdx.x;
  int k = i & 15;
  L[i] = cl[k] * asinhf(2.0f * innerW[i] / cd[k]);
}

// ---- k5: out = A @ logits via fp32 WMMA 16x16x4 ---------------------------

__global__ void __launch_bounds__(256) k5_out(
    const float* __restrict__ A, const float* __restrict__ L,
    float* __restrict__ OUT) {
  const int wave = threadIdx.x >> 5;
  const int lane = threadIdx.x & 31;
  const int hi   = lane >> 4;
  const int ln   = lane & 15;
  const int m0   = blockIdx.x * 128 + wave * 16;

  const float* arow = A + (long)(m0 + ln) * N_ROWS;

  v8f acc = 0.0f;
  for (int k0 = 0; k0 < N_ROWS; k0 += 4) {
    // A 16x4 f32: lane<16 K={0,1}, lane>=16 K={2,3}; streamed -> NT
    v2f av = __builtin_nontemporal_load((const v2f*)(arow + k0 + hi * 2));
    // B 4x16 f32: lane = column, V0/V1 = two K rows (logits stay in L2)
    v2f bv;
    bv[0] = L[(k0 + hi * 2) * NCLS + ln];
    bv[1] = L[(k0 + hi * 2 + 1) * NCLS + ln];
    if ((k0 & 31) == 0 && k0 + 2048 < N_ROWS)
      __builtin_prefetch(arow + k0 + 2048 + hi * 2, 0, 0);
    acc = __builtin_amdgcn_wmma_f32_16x16x4_f32(
        false, av, false, bv, (short)0, acc, false, false);
  }
  #pragma unroll
  for (int j = 0; j < 8; ++j)
    OUT[(long)(m0 + hi * 8 + j) * NCLS + ln] = acc[j];
}

// ---- host launcher --------------------------------------------------------

extern "C" void kernel_launch(void* const* d_in, const int* in_sizes, int n_in,
                              void* d_out, int out_size, void* d_ws, size_t ws_size,
                              hipStream_t stream) {
  (void)in_sizes; (void)n_in; (void)out_size; (void)ws_size;
  const float* X  = (const float*)d_in[0];   // (8192, 256)
  const float* A  = (const float*)d_in[1];   // (8192, 8192)
  const float* W  = (const float*)d_in[2];   // (256, 256)
  const float* Wl = (const float*)d_in[3];   // (256, 16)
  const float* P  = (const float*)d_in[4];   // (16, 256)
  float* out = (float*)d_out;                // (8192, 16)

  char* ws = (char*)d_ws;
  unsigned short* WbT   = (unsigned short*)(ws);                       // 128 KB
  unsigned short* Yt    = (unsigned short*)(ws + 131072);              // 272x8192 bf16
  float*          O272  = (float*)(ws + 131072 + 4456448);             // 8192x272 f32
  float*          innerW= (float*)(ws + 4587520 + 8912896);            // 8192x16 f32
  float*          L     = (float*)(ws + 13500416 + 524288);            // 8192x16 f32
  float*          zpart = (float*)(ws + 14024704 + 524288);            // 128x16 f32
  float*          cl    = (float*)(ws + 14548992 + 8192);              // 16 f32
  float*          cd    = cl + 16;

  k1w_transpose<<<256, 256, 0, stream>>>(W, WbT);
  k1_mobius_matvec<<<64, 256, 0, stream>>>(X, WbT, Yt);
  k2_aggregate<<<64, 256, 0, stream>>>(A, Yt, O272);
  k3_rowops<<<K3_BLKS, 256, 0, stream>>>(O272, P, Wl, innerW, zpart);
  k4a_consts<<<1, 32, 0, stream>>>(Wl, P, zpart, cl, cd);
  k4b_logits<<<512, 256, 0, stream>>>(innerW, cl, cd, L);
  k5_out<<<64, 256, 0, stream>>>(A, L, out);
}
